// WindowAttention4D_41377714929807
// MI455X (gfx1250) — matrix operand order repaired and checked by reading, not compiled, not run on hardware
//
#include <hip/hip_runtime.h>
#include <hip/hip_bf16.h>

typedef __attribute__((ext_vector_type(16))) _Float16 v16h;
typedef __attribute__((ext_vector_type(8)))  _Float16 v8h;
typedef __attribute__((ext_vector_type(8)))  float    v8f;

union AFrag { v16h v; v8h h2[2]; };

__device__ __forceinline__ v8f wmma_f16(v16h a, v16h b, v8f c) {
    // D = A(16x32 f16) * B(32x16 f16) + C(16x16 f32)
    return __builtin_amdgcn_wmma_f32_16x16x32_f16(false, a, false, b, (short)0, c, false, false);
}

// ---------------------------------------------------------------------------
// Kernel 1: QKV projection.  C[131072 x 576] = X[131072 x 192] @ Wqkv^T
// Block tile 128(M) x 64(N), 8 waves, K-steps of 32, double-buffered LDS.
// Epilogue scatters into Q/K/V f16 buffers laid out [b][h][n][d]; Q pre-scaled.
// ---------------------------------------------------------------------------
__global__ __launch_bounds__(256) void qkv_kernel(const float* __restrict__ x,
                                                  const float* __restrict__ Wqkv,
                                                  _Float16* __restrict__ Qh,
                                                  _Float16* __restrict__ Kh,
                                                  _Float16* __restrict__ Vh) {
    __shared__ __align__(32) _Float16 Xs[2][128 * 40];  // [row][k], pad to 40 halves
    __shared__ __align__(32) _Float16 Ws[2][64 * 40];   // [n][k]

    const int tid  = threadIdx.x;
    const int wave = tid >> 5;
    const int lane = tid & 31;
    const int hi   = lane >> 4;
    const int lrow = lane & 15;
    const int m0 = blockIdx.x * 128;
    const int n0 = blockIdx.y * 64;

    // Batched tile loaders: issue all global loads, then convert+store to LDS.
    auto load_tiles = [&](int k0, int buf) {
        float4 fx[4], fw[2];
#pragma unroll
        for (int u = 0; u < 4; ++u) {
            int i = tid + u * 256, r = i >> 3, c4 = i & 7;
            fx[u] = ((const float4*)(x + (size_t)(m0 + r) * 192 + k0))[c4];
        }
#pragma unroll
        for (int u = 0; u < 2; ++u) {
            int i = tid + u * 256, r = i >> 3, c4 = i & 7;
            fw[u] = ((const float4*)(Wqkv + (size_t)(n0 + r) * 192 + k0))[c4];
        }
#pragma unroll
        for (int u = 0; u < 4; ++u) {
            int i = tid + u * 256, r = i >> 3, c4 = i & 7;
            _Float16* dst = &Xs[buf][r * 40 + c4 * 4];
            dst[0] = (_Float16)fx[u].x; dst[1] = (_Float16)fx[u].y;
            dst[2] = (_Float16)fx[u].z; dst[3] = (_Float16)fx[u].w;
        }
#pragma unroll
        for (int u = 0; u < 2; ++u) {
            int i = tid + u * 256, r = i >> 3, c4 = i & 7;
            _Float16* dst = &Ws[buf][r * 40 + c4 * 4];
            dst[0] = (_Float16)fw[u].x; dst[1] = (_Float16)fw[u].y;
            dst[2] = (_Float16)fw[u].z; dst[3] = (_Float16)fw[u].w;
        }
    };

    v8f acc[4] = {};
    load_tiles(0, 0);

    for (int ks = 0; ks < 6; ++ks) {
        const int buf = ks & 1;
        __syncthreads();                       // tile `ks` stores visible
        if (ks < 5) load_tiles((ks + 1) * 32, buf ^ 1);  // overlap next tile

        // Gather A and all four B fragments first so the ds_loads pipeline
        // with partial dscnt waits instead of serializing per-WMMA.
        AFrag au, bu[4];
        const _Float16* arow = &Xs[buf][(wave * 16 + lrow) * 40];
        au.h2[0] = *(const v8h*)(arow + hi * 8);
        au.h2[1] = *(const v8h*)(arow + 16 + hi * 8);
#pragma unroll
        for (int ni = 0; ni < 4; ++ni) {
            const _Float16* brow = &Ws[buf][(ni * 16 + lrow) * 40];
            bu[ni].h2[0] = *(const v8h*)(brow + hi * 8);
            bu[ni].h2[1] = *(const v8h*)(brow + 16 + hi * 8);
        }
#pragma unroll
        for (int ni = 0; ni < 4; ++ni)
            acc[ni] = wmma_f16(au.v, bu[ni].v, acc[ni]);
    }

    const float qscale = 0.17677669529663687f;  // 32^-0.5
#pragma unroll
    for (int ni = 0; ni < 4; ++ni) {
        int j   = n0 + ni * 16 + lrow;   // 0..575
        int s   = j / 192;               // 0=q 1=k 2=v
        int rem = j - s * 192;
        int hh  = rem >> 5;
        int dd  = rem & 31;
        _Float16* base = (s == 0) ? Qh : (s == 1 ? Kh : Vh);
        float sc = (s == 0) ? qscale : 1.0f;
#pragma unroll
        for (int r = 0; r < 8; ++r) {
            int m  = m0 + wave * 16 + r + 8 * hi;
            int b  = m >> 8;
            int nn = m & 255;
            size_t idx = (((size_t)(b * 6 + hh) * 256) + nn) * 32 + dd;
            base[idx] = (_Float16)(acc[ni][r] * sc);
        }
    }
}

// ---------------------------------------------------------------------------
// Kernel 2: flash attention per (window b, head h). 8 waves x 32 query rows.
// K fragments load straight from global; V chunks double-buffered in LDS
// (transposed); S = Q@K^T seeded with the additive mask via WMMA's C operand;
// online softmax with 16-lane shuffle reductions; P restaged through
// wave-private LDS (no barrier needed); one barrier per 32-token chunk.
// ---------------------------------------------------------------------------
__global__ __launch_bounds__(256) void attn_kernel(const _Float16* __restrict__ Qh,
                                                   const _Float16* __restrict__ Kh,
                                                   const _Float16* __restrict__ Vh,
                                                   const float* __restrict__ mask,
                                                   _Float16* __restrict__ Oh) {
    const int bh = blockIdx.x;
    const int b  = bh / 6;
    const int h  = bh % 6;
    const int w  = b & 63;               // mask window index = b % nw
    const int tid  = threadIdx.x;
    const int wave = tid >> 5;
    const int lane = tid & 31;
    const int hi   = lane >> 4;
    const int lrow = lane & 15;

    __shared__ __align__(32) _Float16 Vs[2][32 * 40];     // transposed: [dd][t]
    __shared__ __align__(32) _Float16 Pst[8][32 * 40];    // per-wave P staging

    const size_t headoff = (size_t)(b * 6 + h) * 256 * 32;
    const _Float16* Qg = Qh + headoff;
    const _Float16* Kg = Kh + headoff;
    const _Float16* Vg = Vh + headoff;

    // Batched V-chunk loader: 4 coalesced 2B loads, then 4 transposed LDS stores
    auto load_v = [&](int j0, int buf) {
        _Float16 tv[4];
#pragma unroll
        for (int u = 0; u < 4; ++u) {
            int i = tid + u * 256, t = i >> 5, dd = i & 31;
            tv[u] = Vg[(size_t)(j0 + t) * 32 + dd];
        }
#pragma unroll
        for (int u = 0; u < 4; ++u) {
            int i = tid + u * 256, t = i >> 5, dd = i & 31;
            Vs[buf][dd * 40 + t] = tv[u];
        }
    };

    // Q A-fragments for this wave's 32 rows (rows contiguous in global)
    AFrag qa[2];
#pragma unroll
    for (int mi = 0; mi < 2; ++mi) {
        const _Float16* qrow = Qg + (size_t)(wave * 32 + mi * 16 + lrow) * 32;
        qa[mi].h2[0] = *(const v8h*)(qrow + hi * 8);
        qa[mi].h2[1] = *(const v8h*)(qrow + 16 + hi * 8);
    }

    float m_st[2][8], l_st[2][8];
    v8f o[2][2] = {};
#pragma unroll
    for (int mi = 0; mi < 2; ++mi)
#pragma unroll
        for (int r = 0; r < 8; ++r) { m_st[mi][r] = -1e30f; l_st[mi][r] = 0.0f; }

    load_v(0, 0);

    for (int jc = 0; jc < 8; ++jc) {
        const int j0 = jc * 32;
        const int buf = jc & 1;

        // K B-fragments straight from global (per-lane two 16B chunks)
        AFrag kb[2];
#pragma unroll
        for (int ni = 0; ni < 2; ++ni) {
            const _Float16* krow = Kg + (size_t)(j0 + ni * 16 + lrow) * 32;
            kb[ni].h2[0] = *(const v8h*)(krow + hi * 8);
            kb[ni].h2[1] = *(const v8h*)(krow + 16 + hi * 8);
        }

        // S tiles, with additive window mask loaded as the WMMA C operand
        v8f s[2][2];
#pragma unroll
        for (int mi = 0; mi < 2; ++mi)
#pragma unroll
            for (int ni = 0; ni < 2; ++ni) {
                const float* mrow = mask
                    + ((size_t)w * 256 + (wave * 32 + mi * 16 + 8 * hi)) * 256
                    + (j0 + ni * 16 + lrow);
                v8f c;
#pragma unroll
                for (int r = 0; r < 8; ++r) c[r] = mrow[(size_t)r * 256];
                s[mi][ni] = wmma_f16(qa[mi].v, kb[ni].v, c);
            }

        // Online softmax update (rows replicated across each 16-lane half)
#pragma unroll
        for (int mi = 0; mi < 2; ++mi) {
#pragma unroll
            for (int r = 0; r < 8; ++r) {
                float smax = fmaxf(s[mi][0][r], s[mi][1][r]);
#pragma unroll
                for (int off = 8; off >= 1; off >>= 1)
                    smax = fmaxf(smax, __shfl_xor(smax, off, 32));
                float m_new = fmaxf(m_st[mi][r], smax);
                float alpha = __expf(m_st[mi][r] - m_new);
                float p0 = __expf(s[mi][0][r] - m_new);
                float p1 = __expf(s[mi][1][r] - m_new);
                s[mi][0][r] = p0;
                s[mi][1][r] = p1;
                float rs = p0 + p1;
#pragma unroll
                for (int off = 8; off >= 1; off >>= 1)
                    rs += __shfl_xor(rs, off, 32);
                l_st[mi][r] = l_st[mi][r] * alpha + rs;
                m_st[mi][r] = m_new;
                o[mi][0][r] *= alpha;
                o[mi][1][r] *= alpha;
            }
        }

        // Stage P (f16) into wave-private LDS; same-wave DS ops are in-order,
        // so no barrier is required for the store->gather below.
        _Float16* P = Pst[wave];
#pragma unroll
        for (int mi = 0; mi < 2; ++mi)
#pragma unroll
            for (int ni = 0; ni < 2; ++ni)
#pragma unroll
                for (int r = 0; r < 8; ++r)
                    P[(mi * 16 + r + 8 * hi) * 40 + ni * 16 + lrow] =
                        (_Float16)s[mi][ni][r];

        __syncthreads();                        // Vs[buf] visible to all waves
        if (jc < 7) load_v(j0 + 32, buf ^ 1);   // overlap next V chunk

        // O += P @ V : gather both P fragments and both V fragments first so
        // the ds_loads pipeline ahead of the WMMA chain.
        AFrag pa[2], vb[2];
#pragma unroll
        for (int mi = 0; mi < 2; ++mi) {
            const _Float16* prow = &P[(mi * 16 + lrow) * 40];
            pa[mi].h2[0] = *(const v8h*)(prow + hi * 8);
            pa[mi].h2[1] = *(const v8h*)(prow + 16 + hi * 8);
        }
#pragma unroll
        for (int di = 0; di < 2; ++di) {
            const _Float16* vrow = &Vs[buf][(di * 16 + lrow) * 40];
            vb[di].h2[0] = *(const v8h*)(vrow + hi * 8);
            vb[di].h2[1] = *(const v8h*)(vrow + 16 + hi * 8);
        }
#pragma unroll
        for (int mi = 0; mi < 2; ++mi)
#pragma unroll
            for (int di = 0; di < 2; ++di)
                o[mi][di] = wmma_f16(pa[mi].v, vb[di].v, o[mi][di]);
    }

    // Normalize and write attention output as f16 in [b][n][c] layout
#pragma unroll
    for (int mi = 0; mi < 2; ++mi)
#pragma unroll
        for (int di = 0; di < 2; ++di)
#pragma unroll
            for (int r = 0; r < 8; ++r) {
                int nn = wave * 32 + mi * 16 + r + 8 * hi;
                int dd = di * 16 + lrow;
                float val = o[mi][di][r] / l_st[mi][r];
                Oh[((size_t)b * 256 + nn) * 192 + h * 32 + dd] = (_Float16)val;
            }
}

// ---------------------------------------------------------------------------
// Kernel 3: output projection. out[131072 x 192] = Oh @ Wproj^T + bproj (f32)
// Same double-buffered structure as kernel 1; A tile is already f16.
// ---------------------------------------------------------------------------
__global__ __launch_bounds__(256) void proj_kernel(const _Float16* __restrict__ Oh,
                                                   const float* __restrict__ Wp,
                                                   const float* __restrict__ bp,
                                                   float* __restrict__ out) {
    __shared__ __align__(32) _Float16 Xs[2][128 * 40];
    __shared__ __align__(32) _Float16 Ws[2][64 * 40];

    const int tid  = threadIdx.x;
    const int wave = tid >> 5;
    const int lane = tid & 31;
    const int hi   = lane >> 4;
    const int lrow = lane & 15;
    const int m0 = blockIdx.x * 128;
    const int n0 = blockIdx.y * 64;

    auto load_tiles = [&](int k0, int buf) {
        v8h fa[2];
        float4 fw[2];
#pragma unroll
        for (int u = 0; u < 2; ++u) {
            int i = tid + u * 256, r = i >> 2, c = i & 3;
            fa[u] = *(const v8h*)(Oh + (size_t)(m0 + r) * 192 + k0 + c * 8);
        }
#pragma unroll
        for (int u = 0; u < 2; ++u) {
            int i = tid + u * 256, r = i >> 3, c4 = i & 7;
            fw[u] = ((const float4*)(Wp + (size_t)(n0 + r) * 192 + k0))[c4];
        }
#pragma unroll
        for (int u = 0; u < 2; ++u) {
            int i = tid + u * 256, r = i >> 2, c = i & 3;
            *(v8h*)(&Xs[buf][r * 40 + c * 8]) = fa[u];
        }
#pragma unroll
        for (int u = 0; u < 2; ++u) {
            int i = tid + u * 256, r = i >> 3, c4 = i & 7;
            _Float16* dst = &Ws[buf][r * 40 + c4 * 4];
            dst[0] = (_Float16)fw[u].x; dst[1] = (_Float16)fw[u].y;
            dst[2] = (_Float16)fw[u].z; dst[3] = (_Float16)fw[u].w;
        }
    };

    v8f acc[4] = {};
    load_tiles(0, 0);

    for (int ks = 0; ks < 6; ++ks) {
        const int buf = ks & 1;
        __syncthreads();
        if (ks < 5) load_tiles((ks + 1) * 32, buf ^ 1);

        AFrag au, bu[4];
        const _Float16* arow = &Xs[buf][(wave * 16 + lrow) * 40];
        au.h2[0] = *(const v8h*)(arow + hi * 8);
        au.h2[1] = *(const v8h*)(arow + 16 + hi * 8);
#pragma unroll
        for (int ni = 0; ni < 4; ++ni) {
            const _Float16* brow = &Ws[buf][(ni * 16 + lrow) * 40];
            bu[ni].h2[0] = *(const v8h*)(brow + hi * 8);
            bu[ni].h2[1] = *(const v8h*)(brow + 16 + hi * 8);
        }
#pragma unroll
        for (int ni = 0; ni < 4; ++ni)
            acc[ni] = wmma_f16(au.v, bu[ni].v, acc[ni]);
    }

#pragma unroll
    for (int ni = 0; ni < 4; ++ni) {
        int j = n0 + ni * 16 + lrow;
        float bias = bp[j];
#pragma unroll
        for (int r = 0; r < 8; ++r) {
            int m = m0 + wave * 16 + r + 8 * hi;
            out[(size_t)m * 192 + j] = acc[ni][r] + bias;
        }
    }
}

// ---------------------------------------------------------------------------
extern "C" void kernel_launch(void* const* d_in, const int* in_sizes, int n_in,
                              void* d_out, int out_size, void* d_ws, size_t ws_size,
                              hipStream_t stream) {
    const float* x     = (const float*)d_in[0];
    const float* mask  = (const float*)d_in[1];
    const float* Wqkv  = (const float*)d_in[2];
    const float* Wproj = (const float*)d_in[3];
    const float* bproj = (const float*)d_in[4];
    float* out = (float*)d_out;

    const size_t SQ = (size_t)512 * 6 * 256 * 32;   // per-tensor f16 element count
    _Float16* Qh = (_Float16*)d_ws;
    _Float16* Kh = Qh + SQ;
    _Float16* Vh = Kh + SQ;
    _Float16* Oh = Vh + SQ;                          // [b*n][192] f16

    qkv_kernel<<<dim3(1024, 9), 256, 0, stream>>>(x, Wqkv, Qh, Kh, Vh);
    attn_kernel<<<dim3(512 * 6), 256, 0, stream>>>(Qh, Kh, Vh, mask, Oh);
    proj_kernel<<<dim3(1024, 3), 256, 0, stream>>>(Oh, Wproj, bproj, out);
}
